// HybridBlock_19945828123014
// MI455X (gfx1250) — compile-verified, matrix operand order checked by reading, not run on hardware
//
#include <hip/hip_runtime.h>

typedef __bf16 bf16_t;
typedef __attribute__((ext_vector_type(16))) __bf16 v16bf;
typedef __attribute__((ext_vector_type(8)))  __bf16 v8bf;
typedef __attribute__((ext_vector_type(8)))  float  v8f;

#define DEV static __device__ __forceinline__

// ---------------------------------------------------------------------------
// WMMA helpers (CDNA5 wave32, 16x16x32 bf16 -> f32)
// A fragment: lane m = row, kbase = 8*(lane>>4); VGPR v in [0,3] -> K = kbase+2v..,
// v in [4,7] -> K = 16+kbase+...  => two contiguous 16-byte chunks at +0 and +16.
// B fragment mirrors A with lane = output column, reading rows of B^T ([N][K]).
// ---------------------------------------------------------------------------
DEV v8f wmma_bf16(v16bf a, v16bf b, v8f c) {
  return __builtin_amdgcn_wmma_f32_16x16x32_bf16(false, a, false, b, (short)0, c,
                                                 false, false);
}

DEV const bf16_t* frag_ptr(const bf16_t* tile, int row0, int stride, int lane) {
  return tile + (size_t)(row0 + (lane & 15)) * stride + ((lane >> 4) << 3);
}

DEV v16bf frag_ld(const bf16_t* p) {
  v8bf lo = *(const v8bf*)(p);
  v8bf hi = *(const v8bf*)(p + 16);
  return __builtin_shufflevector(lo, hi, 0, 1, 2, 3, 4, 5, 6, 7,
                                 8, 9, 10, 11, 12, 13, 14, 15);
}

DEV v8f vzero8() {
  v8f z;
#pragma unroll
  for (int i = 0; i < 8; ++i) z[i] = 0.0f;
  return z;
}

DEV float gelu_exact(float x) {
  return 0.5f * x * (1.0f + erff(x * 0.70710678118654752f));
}

// ---------------------------------------------------------------------------
// CDNA5 async global->LDS copy (ASYNCcnt-tracked), 16 bytes per lane.
// lds_byte: DS-space byte offset of the destination (truncated generic addr
// of a __shared__ pointer == addrspace(3) offset).
// ---------------------------------------------------------------------------
DEV uint32_t lds_addr32(const void* p) { return (uint32_t)(uintptr_t)p; }

DEV void async_copy_b128(uint32_t lds_byte, const void* gsrc) {
  asm volatile("global_load_async_to_lds_b128 %0, %1, off"
               :
               : "v"(lds_byte), "v"(gsrc)
               : "memory");
}

DEV void wait_async0() { asm volatile("s_wait_asynccnt 0x0" ::: "memory"); }

// ---------------------------------------------------------------------------
// LayerNorm over D=1024, writes bf16. One block per row, 256 threads x 4 elems.
// ---------------------------------------------------------------------------
__global__ __launch_bounds__(256) void ln_kernel(const float* __restrict__ x,
                                                 const float* __restrict__ g,
                                                 const float* __restrict__ b,
                                                 bf16_t* __restrict__ out) {
  const int row = blockIdx.x;
  const int tid = threadIdx.x, lane = tid & 31, wave = tid >> 5;
  const float4 v = ((const float4*)(x + (size_t)row * 1024))[tid];
  float s = v.x + v.y + v.z + v.w;
  float ss = v.x * v.x + v.y * v.y + v.z * v.z + v.w * v.w;
#pragma unroll
  for (int o = 16; o; o >>= 1) {
    s += __shfl_xor(s, o);
    ss += __shfl_xor(ss, o);
  }
  __shared__ float red[16];
  if (lane == 0) { red[wave] = s; red[8 + wave] = ss; }
  __syncthreads();
  if (tid == 0) {
    float a = 0.f, c = 0.f;
#pragma unroll
    for (int i = 0; i < 8; ++i) { a += red[i]; c += red[8 + i]; }
    red[0] = a; red[8] = c;
  }
  __syncthreads();
  const float mean = red[0] * (1.0f / 1024.0f);
  const float var = red[8] * (1.0f / 1024.0f) - mean * mean;
  const float inv = rsqrtf(var + 1e-5f);
  const float4 gg = ((const float4*)g)[tid];
  const float4 bb = ((const float4*)b)[tid];
  bf16_t* o4 = out + (size_t)row * 1024 + tid * 4;
  o4[0] = (bf16_t)((v.x - mean) * inv * gg.x + bb.x);
  o4[1] = (bf16_t)((v.y - mean) * inv * gg.y + bb.y);
  o4[2] = (bf16_t)((v.z - mean) * inv * gg.z + bb.z);
  o4[3] = (bf16_t)((v.w - mean) * inv * gg.w + bb.w);
}

// ---------------------------------------------------------------------------
// Weight convert+transpose: f32 W[K][N] -> bf16 Wt[N][K]. 32x32 tiles.
// ---------------------------------------------------------------------------
__global__ __launch_bounds__(256) void wt_transpose_kernel(
    const float* __restrict__ W, bf16_t* __restrict__ Wt, int K, int N) {
  __shared__ float t[32][33];
  const int n0 = blockIdx.x * 32, k0 = blockIdx.y * 32;
  const int tx = threadIdx.x & 31, ty = threadIdx.x >> 5;  // 32 x 8
#pragma unroll
  for (int i = 0; i < 32; i += 8)
    t[ty + i][tx] = W[(size_t)(k0 + ty + i) * N + n0 + tx];
  __syncthreads();
#pragma unroll
  for (int i = 0; i < 32; i += 8)
    Wt[(size_t)(n0 + ty + i) * K + k0 + tx] = (bf16_t)t[tx][ty + i];
}

// ---------------------------------------------------------------------------
// Generic bf16 GEMM, f32 accumulate: out[M][N] = A[M][K] * Bt[N][K]^T + bias
// Block: 256 thr = 8 waves; tile 128x128; wave tile 64x32 (4 M-frags x 2 N-frags).
// Double-buffered LDS staged via async global->LDS (one barrier per K-step).
// Epilogue: optional GELU, optional f32 residual, bf16 or f32 output.
// M,N multiples of 128; K multiple of 32.
// ---------------------------------------------------------------------------
template <bool OUT_BF16, bool GELU_ACT, bool RES>
__global__ __launch_bounds__(256) void gemm_kernel(
    const bf16_t* __restrict__ A, const bf16_t* __restrict__ Bt,
    const float* __restrict__ bias, const float* __restrict__ res,
    void* __restrict__ out, int M, int N, int K) {
  const int LDK = 40;  // 32 + 8 pad; 80B row stride -> conflict-free frag reads
  __shared__ __align__(16) bf16_t sA[2][128 * 40];
  __shared__ __align__(16) bf16_t sB[2][128 * 40];
  const int tid = threadIdx.x, lane = tid & 31, wave = tid >> 5;
  const int m0 = blockIdx.y * 128, n0 = blockIdx.x * 128;
  const int wm = (wave >> 2) * 64;  // wave M offset within tile
  const int wn = (wave & 3) * 32;   // wave N offset within tile

  v8f acc[4][2];
#pragma unroll
  for (int i = 0; i < 4; ++i)
#pragma unroll
    for (int j = 0; j < 2; ++j) acc[i][j] = vzero8();

  const int lr = tid >> 1;        // 0..127 (row of both tiles)
  const int lc = (tid & 1) * 16;  // 0 or 16 (16-element half-row)
  const uint32_t ldsA0 = lds_addr32(&sA[0][lr * LDK + lc]);
  const uint32_t ldsB0 = lds_addr32(&sB[0][lr * LDK + lc]);
  const uint32_t bufstride = (uint32_t)(128 * 40 * sizeof(bf16_t));
  const bf16_t* gA = A + (size_t)(m0 + lr) * K + lc;
  const bf16_t* gB = Bt + (size_t)(n0 + lr) * K + lc;

  // stage k0 into buffer `buf` (4 x 16B per thread, ASYNCcnt-tracked)
  auto prefetch = [&](int k0, int buf) {
    const uint32_t la = ldsA0 + (uint32_t)buf * bufstride;
    const uint32_t lb = ldsB0 + (uint32_t)buf * bufstride;
    async_copy_b128(la, gA + k0);
    async_copy_b128(la + 16, gA + k0 + 8);
    async_copy_b128(lb, gB + k0);
    async_copy_b128(lb + 16, gB + k0 + 8);
  };

  prefetch(0, 0);
  const int niter = K >> 5;
  for (int it = 0; it < niter; ++it) {
    const int buf = it & 1;
    wait_async0();      // own async copies (issued last iter) have landed
    __syncthreads();    // everyone's copies landed; everyone done reading buf^1
    if (it + 1 < niter) prefetch((it + 1) << 5, buf ^ 1);
    const bf16_t* tA = sA[buf];
    const bf16_t* tB = sB[buf];
    v16bf af[4], bfv[2];
#pragma unroll
    for (int mt = 0; mt < 4; ++mt)
      af[mt] = frag_ld(frag_ptr(tA, wm + mt * 16, LDK, lane));
#pragma unroll
    for (int nt = 0; nt < 2; ++nt)
      bfv[nt] = frag_ld(frag_ptr(tB, wn + nt * 16, LDK, lane));
#pragma unroll
    for (int mt = 0; mt < 4; ++mt)
#pragma unroll
      for (int nt = 0; nt < 2; ++nt)
        acc[mt][nt] = wmma_bf16(af[mt], bfv[nt], acc[mt][nt]);
  }

  const int cl = lane & 15, rh = (lane >> 4) << 3;
#pragma unroll
  for (int mt = 0; mt < 4; ++mt)
#pragma unroll
    for (int nt = 0; nt < 2; ++nt)
#pragma unroll
      for (int r = 0; r < 8; ++r) {
        const size_t row = (size_t)(m0 + wm + mt * 16 + rh + r);
        const int col = n0 + wn + nt * 16 + cl;
        float v = acc[mt][nt][r] + bias[col];
        if (RES) v += res[row * (size_t)N + col];
        if (GELU_ACT) v = gelu_exact(v);
        if (OUT_BF16)
          ((bf16_t*)out)[row * (size_t)N + col] = (bf16_t)v;
        else
          ((float*)out)[row * (size_t)N + col] = v;
      }
}

// ---------------------------------------------------------------------------
// Sliding-window attention. One block per (query-block, head, batch).
// Q 128x64, K/V context 256x64 (prev block + own block), W=128 band mask.
// All matmuls via WMMA; softmax in registers with 16-lane shuffles.
// qkv: bf16 [B*L][3*1024]; out: bf16 [B*L][1024].
// ---------------------------------------------------------------------------
__global__ __launch_bounds__(256) void swa_kernel(const bf16_t* __restrict__ qkv,
                                                  bf16_t* __restrict__ out) {
  const int blk = blockIdx.x, h = blockIdx.y, b = blockIdx.z;
  const int tid = threadIdx.x, lane = tid & 31, wave = tid >> 5;

  // Q(128x72) + K(256x72) overlap with P(128x264) -- disjoint lifetimes.
  __shared__ __align__(16) char smem_u[128 * 264 * 2];
  __shared__ __align__(16) bf16_t sVt[64 * 264];  // V^T [e][s]
  bf16_t* sQ = (bf16_t*)smem_u;                   // [128][72]
  bf16_t* sK = (bf16_t*)smem_u + 128 * 72;        // [256][72]
  bf16_t* sP = (bf16_t*)smem_u;                   // [128][264]

  const size_t rowQ0 = (size_t)b * 4096 + (size_t)blk * 128;

  // Load Q: thread -> (row tid/2, 32-col half)
  {
    const int r = tid >> 1, half = (tid & 1) * 32;
    const bf16_t* src = qkv + (rowQ0 + r) * 3072 + h * 64 + half;
#pragma unroll
    for (int j = 0; j < 32; j += 8)
      *(v8bf*)&sQ[r * 72 + half + j] = *(const v8bf*)(src + j);
  }
  // Load K and V^T: thread -> key index s = tid (0..255)
  {
    const int s = tid;
    const long lk = (long)blk * 128 - 128 + s;
    if (lk < 0) {
      for (int e = 0; e < 64; ++e) {
        sK[s * 72 + e] = (bf16_t)0.f;
        sVt[e * 264 + s] = (bf16_t)0.f;
      }
    } else {
      const bf16_t* srck = qkv + ((size_t)b * 4096 + lk) * 3072 + 1024 + h * 64;
      const bf16_t* srcv = srck + 1024;
#pragma unroll
      for (int e = 0; e < 64; e += 8) {
        v8bf kv = *(const v8bf*)(srck + e);
        *(v8bf*)&sK[s * 72 + e] = kv;
        v8bf vv = *(const v8bf*)(srcv + e);
#pragma unroll
        for (int j = 0; j < 8; ++j) sVt[(e + j) * 264 + s] = vv[j];
      }
    }
  }
  __syncthreads();

  // Scores: wave w owns query rows [16w,16w+16); 16 key tiles; K-dim = 64.
  v8f acc[16];
#pragma unroll
  for (int i = 0; i < 16; ++i) acc[i] = vzero8();
#pragma unroll
  for (int kk = 0; kk < 64; kk += 32) {
    v16bf aq = frag_ld(frag_ptr(sQ, wave * 16, 72, lane) + kk);
#pragma unroll
    for (int nt = 0; nt < 16; ++nt) {
      v16bf bk = frag_ld(frag_ptr(sK, nt * 16, 72, lane) + kk);
      acc[nt] = wmma_bf16(aq, bk, acc[nt]);
    }
  }

  // Mask + softmax. C-layout: row = 16*wave + 8*(lane>>4) + r, col = 16*nt + (lane&15)
  const int cl = lane & 15, rh = (lane >> 4) << 3;
#pragma unroll
  for (int r = 0; r < 8; ++r) {
    const int rloc = wave * 16 + rh + r;
    float mx = -1e30f;
#pragma unroll
    for (int nt = 0; nt < 16; ++nt) {
      const int s = nt * 16 + cl;
      const bool ok = (s >= rloc + 1) && (s <= rloc + 128) && (blk > 0 || s >= 128);
      const float v = ok ? acc[nt][r] * 0.125f : -1e30f;
      acc[nt][r] = v;
      mx = fmaxf(mx, v);
    }
#pragma unroll
    for (int o = 1; o < 16; o <<= 1) mx = fmaxf(mx, __shfl_xor(mx, o));
    float sum = 0.f;
#pragma unroll
    for (int nt = 0; nt < 16; ++nt) {
      const float e = __expf(acc[nt][r] - mx);
      acc[nt][r] = e;
      sum += e;
    }
#pragma unroll
    for (int o = 1; o < 16; o <<= 1) sum += __shfl_xor(sum, o);
    const float inv = 1.0f / sum;
#pragma unroll
    for (int nt = 0; nt < 16; ++nt) acc[nt][r] *= inv;
  }

  __syncthreads();  // all waves done reading sQ/sK before P overwrites them
#pragma unroll
  for (int nt = 0; nt < 16; ++nt)
#pragma unroll
    for (int r = 0; r < 8; ++r)
      sP[(wave * 16 + rh + r) * 264 + nt * 16 + cl] = (bf16_t)acc[nt][r];
  __syncthreads();

  // PV: out 16x64 per wave; K-dim = 256 keys.
  v8f ov[4];
#pragma unroll
  for (int i = 0; i < 4; ++i) ov[i] = vzero8();
#pragma unroll
  for (int kk = 0; kk < 256; kk += 32) {
    v16bf ap = frag_ld(frag_ptr(sP, wave * 16, 264, lane) + kk);
#pragma unroll
    for (int nt = 0; nt < 4; ++nt) {
      v16bf bv = frag_ld(frag_ptr(sVt, nt * 16, 264, lane) + kk);
      ov[nt] = wmma_bf16(ap, bv, ov[nt]);
    }
  }
#pragma unroll
  for (int nt = 0; nt < 4; ++nt)
#pragma unroll
    for (int r = 0; r < 8; ++r) {
      const size_t row = rowQ0 + wave * 16 + rh + r;
      out[row * 1024 + h * 64 + nt * 16 + cl] = (bf16_t)ov[nt][r];
    }
}

// ---------------------------------------------------------------------------
// Chunked causal linear attention. One block per (b,h); sequential 32-chunk scan.
// State S^T (64x64 f32) + z (64) live in LDS across chunks.
// qkv: f32 [B*L][3*1024] (elu applied at load); out: bf16 [B*L][1024].
// ---------------------------------------------------------------------------
__global__ __launch_bounds__(256) void linattn_kernel(const float* __restrict__ qkv,
                                                      bf16_t* __restrict__ out) {
  const int bh = blockIdx.x, b = bh >> 4, h = bh & 15;
  const int tid = threadIdx.x, lane = tid & 31, wave = tid >> 5;
  __shared__ __align__(16) bf16_t sq[128 * 72];    // q' rows  [c][e]
  __shared__ __align__(16) bf16_t sk[128 * 72];    // k' rows  [c][e]
  __shared__ __align__(16) bf16_t skT[64 * 136];   // k'^T     [e][c]
  __shared__ __align__(16) bf16_t svT[64 * 136];   // v^T      [f][c]
  __shared__ __align__(16) bf16_t sP[128 * 136];   // masked A [c][d]
  __shared__ __align__(16) bf16_t sSB[64 * 72];    // bf16 snapshot of S^T [f][e]
  __shared__ float sST[64 * 64];                   // S^T f32  [f][e]
  __shared__ float sz[64];
  __shared__ float srow[128];
  __shared__ float sden[128];

  for (int i = tid; i < 64 * 64; i += 256) sST[i] = 0.f;
  if (tid < 64) sz[tid] = 0.f;
  __syncthreads();

  const int cl = lane & 15, rh = (lane >> 4) << 3;

  for (int ch = 0; ch < 32; ++ch) {
    // 1. snapshot S^T -> bf16 (B-operand layout: row f, cols e)
    for (int i = tid; i < 64 * 64; i += 256)
      sSB[(i >> 6) * 72 + (i & 63)] = (bf16_t)sST[i];
    // 2. load chunk, apply elu maps
    {
      const int c = tid >> 1, half = (tid & 1) * 32;
      const size_t row = (size_t)b * 4096 + (size_t)ch * 128 + c;
      const float* pq = qkv + row * 3072 + h * 64 + half;
      const float* pk = pq + 1024;
      const float* pv = pq + 2048;
#pragma unroll 8
      for (int j = 0; j < 32; ++j) {
        float qv = pq[j] * 0.125f;  // E^-0.5
        qv = (qv > 0.f ? qv : expm1f(qv)) + 1.0001f;
        sq[c * 72 + half + j] = (bf16_t)qv;
        float kv = pk[j];
        kv = (kv > 0.f ? kv : expm1f(kv)) + 1.0001f;
        const bf16_t kb = (bf16_t)kv;
        sk[c * 72 + half + j] = kb;
        skT[(half + j) * 136 + c] = kb;
        svT[(half + j) * 136 + c] = (bf16_t)pv[j];
      }
    }
    __syncthreads();

    // 3. A = q k'^T (8 tiles/wave) and num1 = q @ S (4 tiles/wave), K-dim 64
    v8f accA[8], accN[4];
#pragma unroll
    for (int i = 0; i < 8; ++i) accA[i] = vzero8();
#pragma unroll
    for (int i = 0; i < 4; ++i) accN[i] = vzero8();
#pragma unroll
    for (int kk = 0; kk < 64; kk += 32) {
      v16bf aq = frag_ld(frag_ptr(sq, wave * 16, 72, lane) + kk);
#pragma unroll
      for (int nt = 0; nt < 8; ++nt) {
        v16bf bk = frag_ld(frag_ptr(sk, nt * 16, 72, lane) + kk);
        accA[nt] = wmma_bf16(aq, bk, accA[nt]);
      }
#pragma unroll
      for (int nt = 0; nt < 4; ++nt) {
        v16bf bs = frag_ld(frag_ptr(sSB, nt * 16, 72, lane) + kk);
        accN[nt] = wmma_bf16(aq, bs, accN[nt]);
      }
    }
    // tril mask + row sums + stage masked A as bf16
#pragma unroll
    for (int r = 0; r < 8; ++r) {
      const int c = wave * 16 + rh + r;
      float rs = 0.f;
#pragma unroll
      for (int nt = 0; nt < 8; ++nt) {
        const int d = nt * 16 + cl;
        const float v = (d <= c) ? accA[nt][r] : 0.f;
        accA[nt][r] = v;
        rs += v;
      }
#pragma unroll
      for (int o = 1; o < 16; o <<= 1) rs += __shfl_xor(rs, o);
      if (cl == 0) srow[c] = rs;
#pragma unroll
      for (int nt = 0; nt < 8; ++nt)
        sP[c * 136 + nt * 16 + cl] = (bf16_t)accA[nt][r];
    }
    __syncthreads();

    // 4. den[c] = q'.z + rowsum(A)   (uses pre-update z)
    if (tid < 128) {
      float qz = 0.f;
#pragma unroll 8
      for (int e = 0; e < 64; ++e) qz += (float)sq[tid * 72 + e] * sz[e];
      sden[tid] = qz + srow[tid];
    }
    __syncthreads();

    // 5. num2 = A @ v  (B-operand = v^T rows), K-dim 128
#pragma unroll
    for (int kk = 0; kk < 128; kk += 32) {
      v16bf ap = frag_ld(frag_ptr(sP, wave * 16, 136, lane) + kk);
#pragma unroll
      for (int nt = 0; nt < 4; ++nt) {
        v16bf bv = frag_ld(frag_ptr(svT, nt * 16, 136, lane) + kk);
        accN[nt] = wmma_bf16(ap, bv, accN[nt]);
      }
    }
    // 6. o = num / (den + 1e-6) -> global bf16
#pragma unroll
    for (int nt = 0; nt < 4; ++nt)
#pragma unroll
      for (int r = 0; r < 8; ++r) {
        const int c = wave * 16 + rh + r;
        const float o = accN[nt][r] / (sden[c] + 1e-6f);
        const size_t row = (size_t)b * 4096 + (size_t)ch * 128 + c;
        out[row * 1024 + h * 64 + nt * 16 + cl] = (bf16_t)o;
      }
    __syncthreads();

    // 7. S^T += v^T @ k'  (16 tiles over 8 waves, 2 each); z += colsum(k')
    {
      const int ft = wave >> 1;
      const int ebase = (wave & 1) * 32;
      v8f accS[2];
#pragma unroll
      for (int i = 0; i < 2; ++i) accS[i] = vzero8();
#pragma unroll
      for (int kk = 0; kk < 128; kk += 32) {
        v16bf av = frag_ld(frag_ptr(svT, ft * 16, 136, lane) + kk);
#pragma unroll
        for (int nt = 0; nt < 2; ++nt) {
          v16bf bk = frag_ld(frag_ptr(skT, ebase + nt * 16, 136, lane) + kk);
          accS[nt] = wmma_bf16(av, bk, accS[nt]);
        }
      }
#pragma unroll
      for (int nt = 0; nt < 2; ++nt)
#pragma unroll
        for (int r = 0; r < 8; ++r) {
          const int f = ft * 16 + rh + r;
          const int e = ebase + nt * 16 + cl;
          sST[f * 64 + e] += accS[nt][r];
        }
    }
    if (tid < 64) {
      float zs = 0.f;
#pragma unroll 8
      for (int c = 0; c < 128; ++c) zs += (float)sk[c * 72 + tid];
      sz[tid] += zs;
    }
    __syncthreads();
  }
}

// ---------------------------------------------------------------------------
// Launch: B=2, L=4096, D=1024, H=16, E=64, W=128. Requires ws >= ~254 MB.
// ---------------------------------------------------------------------------
extern "C" void kernel_launch(void* const* d_in, const int* in_sizes, int n_in,
                              void* d_out, int out_size, void* d_ws, size_t ws_size,
                              hipStream_t stream) {
  const float* x = (const float*)d_in[0];
  const float* g1 = (const float*)d_in[1];
  const float* b1 = (const float*)d_in[2];
  const float* Wqkva = (const float*)d_in[3];
  const float* bqkva = (const float*)d_in[4];
  const float* Wpa = (const float*)d_in[5];
  const float* bpa = (const float*)d_in[6];
  const float* Wqkvs = (const float*)d_in[7];
  const float* bqkvs = (const float*)d_in[8];
  const float* Wos = (const float*)d_in[9];
  const float* bos = (const float*)d_in[10];
  const float* Wo = (const float*)d_in[11];
  const float* bo = (const float*)d_in[12];
  const float* gm = (const float*)d_in[13];
  const float* bm = (const float*)d_in[14];
  const float* W1 = (const float*)d_in[15];
  const float* b1m = (const float*)d_in[16];
  const float* W2 = (const float*)d_in[17];
  const float* b2m = (const float*)d_in[18];

  char* ws = (char*)d_ws;
  bf16_t* WqkvaT = (bf16_t*)(ws + 0);            // [3072][1024]
  bf16_t* WqkvsT = (bf16_t*)(ws + 6291456);      // [3072][1024]
  bf16_t* WpaT = (bf16_t*)(ws + 12582912);       // [1024][1024]
  bf16_t* WosT = (bf16_t*)(ws + 14680064);       // [1024][1024]
  bf16_t* WoT = (bf16_t*)(ws + 16777216);        // [1024][1024]
  bf16_t* W1T = (bf16_t*)(ws + 18874368);        // [4096][1024]
  bf16_t* W2T = (bf16_t*)(ws + 27262976);        // [1024][4096]
  bf16_t* xn = (bf16_t*)(ws + 35651584);         // [8192][1024]  (later: h)
  bf16_t* swa_o = (bf16_t*)(ws + 52428800);      // [8192][1024]
  bf16_t* lin_o = (bf16_t*)(ws + 69206016);      // [8192][1024]
  bf16_t* mix = (bf16_t*)(ws + 85983232);        // [8192][1024]
  bf16_t* qkva = (bf16_t*)(ws + 102760448);      // [8192][3072]
  float* qkvs = (float*)(ws + 153092096);        // [8192][3072]
  float* attnf = (float*)(ws + 153092096);       // [8192][1024] (reuses qkvs)
  bf16_t* gbuf = (bf16_t*)(ws + 153092096);      // [8192][4096] (reuses attnf)
  float* x1 = (float*)(ws + 220200960);          // [8192][1024] (tail of qkvs)
  float* outf = (float*)d_out;

  const dim3 blk(256);
  // Weight convert+transpose to bf16 [N][K]
  wt_transpose_kernel<<<dim3(3072 / 32, 1024 / 32), blk, 0, stream>>>(Wqkva, WqkvaT, 1024, 3072);
  wt_transpose_kernel<<<dim3(3072 / 32, 1024 / 32), blk, 0, stream>>>(Wqkvs, WqkvsT, 1024, 3072);
  wt_transpose_kernel<<<dim3(1024 / 32, 1024 / 32), blk, 0, stream>>>(Wpa, WpaT, 1024, 1024);
  wt_transpose_kernel<<<dim3(1024 / 32, 1024 / 32), blk, 0, stream>>>(Wos, WosT, 1024, 1024);
  wt_transpose_kernel<<<dim3(1024 / 32, 1024 / 32), blk, 0, stream>>>(Wo, WoT, 1024, 1024);
  wt_transpose_kernel<<<dim3(4096 / 32, 1024 / 32), blk, 0, stream>>>(W1, W1T, 1024, 4096);
  wt_transpose_kernel<<<dim3(1024 / 32, 4096 / 32), blk, 0, stream>>>(W2, W2T, 4096, 1024);

  // xn = LN(x)
  ln_kernel<<<8192, blk, 0, stream>>>(x, g1, b1, xn);
  // qkv projections
  gemm_kernel<true, false, false><<<dim3(24, 64), blk, 0, stream>>>(
      xn, WqkvaT, bqkva, nullptr, qkva, 8192, 3072, 1024);
  gemm_kernel<false, false, false><<<dim3(24, 64), blk, 0, stream>>>(
      xn, WqkvsT, bqkvs, nullptr, qkvs, 8192, 3072, 1024);
  // attention branches
  swa_kernel<<<dim3(32, 16, 2), blk, 0, stream>>>(qkva, swa_o);
  linattn_kernel<<<dim3(32), blk, 0, stream>>>(qkvs, lin_o);
  // attn_out = swa @ Wp + bp  (f32)
  gemm_kernel<false, false, false><<<dim3(8, 64), blk, 0, stream>>>(
      swa_o, WpaT, bpa, nullptr, attnf, 8192, 1024, 1024);
  // mix = lin @ Wo_s + bo_s + attn_out  (bf16)
  gemm_kernel<true, false, true><<<dim3(8, 64), blk, 0, stream>>>(
      lin_o, WosT, bos, attnf, mix, 8192, 1024, 1024);
  // x1 = mix @ Wo + bo + x  (f32)
  gemm_kernel<false, false, true><<<dim3(8, 64), blk, 0, stream>>>(
      mix, WoT, bo, x, x1, 8192, 1024, 1024);
  // h = LN(x1)
  ln_kernel<<<8192, blk, 0, stream>>>(x1, gm, bm, xn);
  // g = gelu(h @ W1 + b1m)  (bf16)
  gemm_kernel<true, true, false><<<dim3(32, 64), blk, 0, stream>>>(
      xn, W1T, b1m, nullptr, gbuf, 8192, 4096, 1024);
  // out = g @ W2 + b2m + x1  (f32)
  gemm_kernel<false, false, true><<<dim3(8, 64), blk, 0, stream>>>(
      gbuf, W2T, b2m, x1, outf, 8192, 1024, 4096);
}
